// HeteroGNN_51702816309361
// MI455X (gfx1250) — matrix operand order repaired and checked by reading, not compiled, not run on hardware
//
#include <hip/hip_runtime.h>
#include <cstdint>

// ---------------------------------------------------------------------------
// HeteroGNN for MI455X (gfx1250, wave32, WMMA).
//  B = 131072 nodes/edges, T = 4, D = 64, H = 2, DK = 32, L = 2.
//  Dense math: v_wmma_f32_16x16x32_f16 (f16 operands, f32 accumulate).
//  Graph scatter: L2 float atomics (max via int punning, add native).
// ---------------------------------------------------------------------------

#define NBN   131072
#define LAYERS 2
#define LNEPS 1e-5f

typedef __attribute__((ext_vector_type(16))) _Float16 v16h;
typedef __attribute__((ext_vector_type(8)))  float    v8f;

__device__ __forceinline__ void atomicMaxF(float* addr, float val) {
  // Sign-split trick: valid total order on IEEE bits; init value is 0xFFFFFFFF.
  if (val >= 0.0f) atomicMax((int*)addr, __float_as_int(val));
  else             atomicMin((unsigned int*)addr, __float_as_uint(val));
}

// ---------- mean over T=4 (the only HBM-heavy pass) -------------------------
__global__ void mean_kernel(const float* __restrict__ X, float* __restrict__ outF,
                            _Float16* __restrict__ outH, float* __restrict__ copyF,
                            float* __restrict__ sumF, int n) {
  int i = blockIdx.x * blockDim.x + threadIdx.x;
  if (i >= n) return;
  int b = i >> 6, d = i & 63;
  const float* p = X + (size_t)b * 256 + d;
  float s = p[0] + p[64] + p[128] + p[192];
  float mn = s * 0.25f;
  if (outF)  outF[i]  = mn;
  if (outH)  outH[i]  = (_Float16)mn;
  if (copyF) copyF[i] = mn;
  if (sumF)  sumF[i]  = s;
}

// ---------- weight pre-pack into WMMA B-operand lane layout -----------------
// Packed element id = ((tileN*2 + kstep)*32 + lane)*16 + e
//   lane<16 : N = lane,    K = kstep*32 + (e&7) + (e>=8 ? 16 : 0)
//   lane>=16: N = lane-16, K = kstep*32 + 8 + (e&7) + (e>=8 ? 16 : 0)
// Optionally folds rel[h,d,f] (einsum) and per-head scale pri[h]*gs.
__global__ void packw_kernel(const float* __restrict__ W, const float* __restrict__ rel,
                             const float* __restrict__ pri, float gs,
                             _Float16* __restrict__ out) {
  int id = blockIdx.x * blockDim.x + threadIdx.x;
  if (id >= 4096) return;
  int e = id & 15, l = (id >> 4) & 31, s = (id >> 9) & 1, t = id >> 10;
  int k = s * 32 + ((l >= 16) ? 8 : 0) + (e & 7) + ((e >= 8) ? 16 : 0);
  int n = t * 16 + (l & 15);
  float v;
  if (rel) {
    int h = n >> 5, f = n & 31;
    float acc = 0.0f;
    for (int dd = 0; dd < 32; ++dd)
      acc += W[k * 64 + h * 32 + dd] * rel[h * 1024 + dd * 32 + f];
    v = acc * gs * (pri ? pri[h] : 1.0f);
  } else {
    v = W[k * 64 + n] * gs;
  }
  out[id] = (_Float16)v;
}

__global__ void packb_kernel(const float* __restrict__ bsrc, const float* __restrict__ rel,
                             const float* __restrict__ pri, float gs,
                             float* __restrict__ out) {
  int n = threadIdx.x;
  if (n >= 64) return;
  int h = n >> 5, f = n & 31;
  float acc = 0.0f;
  for (int dd = 0; dd < 32; ++dd)
    acc += bsrc[h * 32 + dd] * rel[h * 1024 + dd * 32 + f];
  out[n] = acc * gs * (pri ? pri[h] : 1.0f);
}

// ---------- 64-wide GEMM: C[B,64] = op(A)[B,64] @ W[64,64] + bias -----------
// One wave = 16 output rows, 8 WMMAs. AMODE 0: A is f16 row-major.
// AMODE 1: A is f32 row-major, exact gelu applied, then f16. RELU: epilogue.
template<int AMODE, int RELU>
__global__ void gemm64_kernel(const _Float16* __restrict__ A16, const float* __restrict__ A32,
                              const _Float16* __restrict__ Bp, const float* __restrict__ bias,
                              float* __restrict__ C, int nrows) {
  int wave = (int)((blockIdx.x * blockDim.x + threadIdx.x) >> 5);
  int lane = threadIdx.x & 31;
  int row0 = wave * 16;
  if (row0 >= nrows) return;            // wave-uniform: EXEC all-1s inside
  int hi = lane >> 4, l16 = lane & 15;
  int off = hi * 8;
  int arow = row0 + l16;
  v16h a0, a1;
  if (AMODE == 0) {
    const _Float16* Ar = A16 + (size_t)arow * 64;
    ((uint4*)&a0)[0] = *(const uint4*)(Ar + off);        // K {0..7}+off
    ((uint4*)&a0)[1] = *(const uint4*)(Ar + 16 + off);   // K {16..23}+off
    ((uint4*)&a1)[0] = *(const uint4*)(Ar + 32 + off);
    ((uint4*)&a1)[1] = *(const uint4*)(Ar + 48 + off);
  } else {
    const float* Ar = A32 + (size_t)arow * 64;
    #pragma unroll
    for (int c4 = 0; c4 < 4; ++c4) {
      #pragma unroll
      for (int j = 0; j < 8; ++j) {
        float x = Ar[c4 * 16 + off + j];
        x = 0.5f * x * (1.0f + erff(x * 0.70710678118f)); // exact gelu
        if (c4 < 2) a0[c4 * 8 + j] = (_Float16)x;
        else        a1[(c4 - 2) * 8 + j] = (_Float16)x;
      }
    }
  }
  #pragma unroll
  for (int t = 0; t < 4; ++t) {
    v16h b0, b1;
    const _Float16* bp0 = Bp + ((size_t)(t * 2 + 0) * 32 + lane) * 16;
    const _Float16* bp1 = Bp + ((size_t)(t * 2 + 1) * 32 + lane) * 16;
    ((uint4*)&b0)[0] = *(const uint4*)(bp0);
    ((uint4*)&b0)[1] = *(const uint4*)(bp0 + 8);
    ((uint4*)&b1)[0] = *(const uint4*)(bp1);
    ((uint4*)&b1)[1] = *(const uint4*)(bp1 + 8);
    v8f c = {};
    c = __builtin_amdgcn_wmma_f32_16x16x32_f16(false, a0, false, b0, (short)0, c, false, false);
    c = __builtin_amdgcn_wmma_f32_16x16x32_f16(false, a1, false, b1, (short)0, c, false, false);
    int ncol = t * 16 + l16;
    float bv = bias ? bias[ncol] : 0.0f;
    #pragma unroll
    for (int j = 0; j < 8; ++j) {
      int r = row0 + j + hi * 8;        // C/D layout: VGPR j -> M = j (+8 hi-half)
      float val = c[j] + bv;
      if (RELU) val = fmaxf(val, 0.0f);
      C[(size_t)r * 64 + ncol] = val;
    }
  }
}

// ---------- edge kernels (L2-resident gathers + atomics) --------------------
__global__ void edge_logit_kernel(const float* __restrict__ Q, const float* __restrict__ K,
                                  const int* __restrict__ ei, float* __restrict__ logit,
                                  float* __restrict__ m, int E) {
  int e = blockIdx.x * blockDim.x + threadIdx.x;
  if (e >= E) return;
  int row = ei[e], col = ei[E + e];
  const float4* q = (const float4*)(Q + (size_t)col * 64);
  const float4* k = (const float4*)(K + (size_t)row * 64);
  float s0 = 0.0f, s1 = 0.0f;
  #pragma unroll
  for (int i = 0; i < 8; ++i) { float4 a = q[i], b = k[i]; s0 += a.x*b.x + a.y*b.y + a.z*b.z + a.w*b.w; }
  #pragma unroll
  for (int i = 8; i < 16; ++i) { float4 a = q[i], b = k[i]; s1 += a.x*b.x + a.y*b.y + a.z*b.z + a.w*b.w; }
  logit[(size_t)e * 2 + 0] = s0;        // rel_pri/sqrt(dk) already folded into K
  logit[(size_t)e * 2 + 1] = s1;
  atomicMaxF(m + (size_t)col * 2 + 0, s0);
  atomicMaxF(m + (size_t)col * 2 + 1, s1);
}

__global__ void edge_exp_kernel(float* __restrict__ exb, const float* __restrict__ m,
                                const int* __restrict__ ei, float* __restrict__ denom, int E) {
  int idx = blockIdx.x * blockDim.x + threadIdx.x;
  if (idx >= E * 2) return;
  int e = idx >> 1, h = idx & 1;
  int col = ei[E + e];
  float v = expf(exb[idx] - m[(size_t)col * 2 + h]);
  exb[idx] = v;
  atomicAdd(denom + (size_t)col * 2 + h, v);
}

__global__ void edge_msg_kernel(const float* __restrict__ exb, const float* __restrict__ denom,
                                const float* __restrict__ V, const float* __restrict__ eae,
                                const int* __restrict__ ei, float* __restrict__ aggr, int E) {
  int idx = blockIdx.x * blockDim.x + threadIdx.x;
  if (idx >= E * 64) return;
  int e = idx >> 6, j = idx & 63, h = j >> 5;
  int col = ei[E + e], row = ei[e];
  float att = exb[(size_t)e * 2 + h] / denom[(size_t)col * 2 + h];
  float val = V[(size_t)row * 64 + j] * att + eae[(size_t)e * 64 + j];
  atomicAdd(aggr + (size_t)col * 64 + j, val);
}

// ---------- residual + LayerNorm + f16 recast + layer-sum accumulate --------
__global__ void ln_kernel(const float* __restrict__ Y0, const float* __restrict__ Xres,
                          const float* __restrict__ g, const float* __restrict__ bb,
                          float* __restrict__ outF, _Float16* __restrict__ outH,
                          float* __restrict__ accum, int n) {
  int r = blockIdx.x * blockDim.x + threadIdx.x;
  if (r >= n) return;
  size_t base = (size_t)r * 64;
  float y[64]; float mu = 0.0f;
  #pragma unroll
  for (int j = 0; j < 64; ++j) { y[j] = Y0[base + j] + Xres[base + j]; mu += y[j]; }
  mu *= (1.0f / 64.0f);
  float var = 0.0f;
  #pragma unroll
  for (int j = 0; j < 64; ++j) { float d = y[j] - mu; var += d * d; }
  float inv = rsqrtf(var * (1.0f / 64.0f) + LNEPS);
  #pragma unroll
  for (int j = 0; j < 64; ++j) {
    float o = (y[j] - mu) * inv * g[j] + bb[j];
    outF[base + j] = o;
    outH[base + j] = (_Float16)o;
    accum[base + j] += o;
  }
}

// ---------- final head ------------------------------------------------------
__global__ void finalcat_kernel(const float* __restrict__ u, const float* __restrict__ it,
                                const float* __restrict__ ea, _Float16* __restrict__ out, int n) {
  int i = blockIdx.x * blockDim.x + threadIdx.x;
  if (i >= n) return;
  out[i] = (_Float16)((u[i] + it[i]) * (1.0f / 3.0f) + ea[i]);
}

__global__ void fc2_kernel(const float* __restrict__ Hf, const float* __restrict__ w2,
                           const float* __restrict__ b2, float* __restrict__ out, int n) {
  int b = blockIdx.x * blockDim.x + threadIdx.x;
  if (b >= n) return;
  const float4* h = (const float4*)(Hf + (size_t)b * 64);
  const float4* w = (const float4*)w2;
  float s = 0.0f;
  #pragma unroll
  for (int i = 0; i < 16; ++i) { float4 a = h[i], ww = w[i]; s += a.x*ww.x + a.y*ww.y + a.z*ww.z + a.w*ww.w; }
  out[b] = s + b2[0];
}

// ---------------------------------------------------------------------------
extern "C" void kernel_launch(void* const* d_in, const int* in_sizes, int n_in,
                              void* d_out, int out_size, void* d_ws, size_t ws_size,
                              hipStream_t stream) {
  (void)in_sizes; (void)n_in; (void)out_size; (void)ws_size;
  const int NB = NBN;
  const size_t U32 = (size_t)NB * 64 * sizeof(float);
  const size_t U16 = (size_t)NB * 64 * sizeof(_Float16);
  const size_t S2  = (size_t)NB * 2 * sizeof(float);

  // Input flattening assumed: recursive insertion order of setup_inputs() dict.
  const float* x_user = (const float*)d_in[0];
  const float* x_item = (const float*)d_in[1];
  const float* ea_ui  = (const float*)d_in[2];
  const float* ea_iu  = (const float*)d_in[3];
  const float* fc1_w  = (const float*)d_in[64];
  const float* fc1_b  = (const float*)d_in[65];
  const float* fc2_w  = (const float*)d_in[66];
  const float* fc2_b  = (const float*)d_in[67];
  const int*   ei_ui  = (const int*)d_in[68];
  const int*   ei_iu  = (const int*)d_in[69];

  struct ConvP {
    const float *w3,*b3,*wq,*bq,*wk,*bk,*wv,*bv,*wa,*ba,*rel_att,*rel_msg,*rel_pri,*ln_g,*ln_b;
  };
  auto getConv = [&](int dir, int l) -> ConvP {
    int b = 4 + (dir * LAYERS + l) * 15;   // 'ui' l0=4, l1=19; 'iu' l0=34, l1=49
    ConvP p;
    p.w3      = (const float*)d_in[b + 0];  p.b3      = (const float*)d_in[b + 1];
    p.wq      = (const float*)d_in[b + 2];  p.bq      = (const float*)d_in[b + 3];
    p.wk      = (const float*)d_in[b + 4];  p.bk      = (const float*)d_in[b + 5];
    p.wv      = (const float*)d_in[b + 6];  p.bv      = (const float*)d_in[b + 7];
    p.wa      = (const float*)d_in[b + 8];  p.ba      = (const float*)d_in[b + 9];
    p.rel_att = (const float*)d_in[b + 10]; p.rel_msg = (const float*)d_in[b + 11];
    p.rel_pri = (const float*)d_in[b + 12]; p.ln_g    = (const float*)d_in[b + 13];
    p.ln_b    = (const float*)d_in[b + 14];
    return p;
  };

  // ---- workspace carve (~560 MB) ----
  size_t off = 0;
  auto walloc = [&](size_t bytes) -> void* {
    void* p = (char*)d_ws + off; off += (bytes + 255) & ~(size_t)255; return p;
  };
  float* xuA  = (float*)walloc(U32);  float* xiA  = (float*)walloc(U32);
  float* xuB  = (float*)walloc(U32);  float* xiB  = (float*)walloc(U32);
  float* usum = (float*)walloc(U32);  float* isum = (float*)walloc(U32);
  float* easum= (float*)walloc(U32);
  float* Qb   = (float*)walloc(U32);  float* Kb   = (float*)walloc(U32);
  float* Vb   = (float*)walloc(U32);  float* EAE  = (float*)walloc(U32);
  float* AGG  = (float*)walloc(U32);  float* Hbuf = (float*)walloc(U32);
  _Float16* xu16A  = (_Float16*)walloc(U16); _Float16* xi16A  = (_Float16*)walloc(U16);
  _Float16* xu16B  = (_Float16*)walloc(U16); _Float16* xi16B  = (_Float16*)walloc(U16);
  _Float16* eamui16= (_Float16*)walloc(U16); _Float16* eamiu16= (_Float16*)walloc(U16);
  _Float16* fin16  = (_Float16*)walloc(U16);
  float* mbuf  = (float*)walloc(S2); float* denom = (float*)walloc(S2);
  float* exbuf = (float*)walloc(S2);
  _Float16* packs = (_Float16*)walloc((size_t)(4 * 5 + 1) * 4096 * sizeof(_Float16));
  float* effb = (float*)walloc((size_t)4 * 2 * 64 * sizeof(float));

  dim3 blk(256);
  int gElem = (NB * 64 + 255) / 256;   // 32768
  int gNode = (NB + 255) / 256;        // 512
  int gE2   = (NB * 2 + 255) / 256;    // 1024
  int gGemm = NB / 128;                // 1024 blocks x 8 waves x 16 rows
  const float invs = 0.17677669529663689f; // 1/sqrt(32)

  // ---- T-mean reductions (only HBM-heavy pass) ----
  mean_kernel<<<gElem, blk, 0, stream>>>(x_user, xuA, xu16A, usum, nullptr, NB * 64);
  mean_kernel<<<gElem, blk, 0, stream>>>(x_item, xiA, xi16A, isum, nullptr, NB * 64);
  mean_kernel<<<gElem, blk, 0, stream>>>(ea_ui, nullptr, eamui16, nullptr, easum, NB * 64);
  mean_kernel<<<gElem, blk, 0, stream>>>(ea_iu, nullptr, eamiu16, nullptr, nullptr, NB * 64);

  // ---- weight packing (rel_att/rel_msg/rel_pri folded into K/V weights) ----
  for (int c = 0; c < 4; ++c) {
    ConvP p = getConv(c >> 1, c & 1);
    _Float16* pk = packs + (size_t)c * 5 * 4096;
    packw_kernel<<<16, blk, 0, stream>>>(p.w3, nullptr,   nullptr,   1.0f, pk + 0 * 4096);
    packw_kernel<<<16, blk, 0, stream>>>(p.wq, nullptr,   nullptr,   1.0f, pk + 1 * 4096);
    packw_kernel<<<16, blk, 0, stream>>>(p.wk, p.rel_att, p.rel_pri, invs, pk + 2 * 4096);
    packw_kernel<<<16, blk, 0, stream>>>(p.wv, p.rel_msg, nullptr,   1.0f, pk + 3 * 4096);
    packw_kernel<<<16, blk, 0, stream>>>(p.wa, nullptr,   nullptr,   1.0f, pk + 4 * 4096);
    packb_kernel<<<1, 64, 0, stream>>>(p.bk, p.rel_att, p.rel_pri, invs, effb + c * 128);
    packb_kernel<<<1, 64, 0, stream>>>(p.bv, p.rel_msg, nullptr,   1.0f, effb + c * 128 + 64);
  }
  packw_kernel<<<16, blk, 0, stream>>>(fc1_w, nullptr, nullptr, 1.0f, packs + 20 * 4096);

  // ---- one bipartite HGT conv ----
  auto runConv = [&](const ConvP& p, int c, _Float16* src16, _Float16* dst16, float* dstF,
                     _Float16* eam16, const int* ei, float* outF, _Float16* outH, float* accum) {
    _Float16* pk = packs + (size_t)c * 5 * 4096;
    float* bkeff = effb + c * 128;
    float* bveff = effb + c * 128 + 64;
    hipMemsetAsync(denom, 0, S2, stream);
    hipMemsetAsync(AGG, 0, U32, stream);
    hipMemsetAsync(mbuf, 0xFF, S2, stream);   // -NaN bits: valid init for atomicMaxF
    gemm64_kernel<0,0><<<gGemm, blk, 0, stream>>>(dst16, nullptr, pk + 1 * 4096, p.bq, Qb, NB);
    gemm64_kernel<0,0><<<gGemm, blk, 0, stream>>>(src16, nullptr, pk + 2 * 4096, bkeff, Kb, NB);
    gemm64_kernel<0,0><<<gGemm, blk, 0, stream>>>(src16, nullptr, pk + 3 * 4096, bveff, Vb, NB);
    gemm64_kernel<0,0><<<gGemm, blk, 0, stream>>>(eam16, nullptr, pk + 0 * 4096, p.b3, EAE, NB);
    edge_logit_kernel<<<gNode, blk, 0, stream>>>(Qb, Kb, ei, exbuf, mbuf, NB);
    edge_exp_kernel<<<gE2, blk, 0, stream>>>(exbuf, mbuf, ei, denom, NB);
    edge_msg_kernel<<<gElem, blk, 0, stream>>>(exbuf, denom, Vb, EAE, ei, AGG, NB);
    // y0 = gelu(aggr) @ wa + ba   (Q buffer reused as y0; Q dead by now)
    gemm64_kernel<1,0><<<gGemm, blk, 0, stream>>>(nullptr, AGG, pk + 4 * 4096, p.ba, Qb, NB);
    ln_kernel<<<gNode, blk, 0, stream>>>(Qb, dstF, p.ln_g, p.ln_b, outF, outH, accum, NB);
  };

  // ---- layer loop (double-buffered: both convs read old xu/xi) ----
  float *xu = xuA, *xi = xiA, *xun = xuB, *xin = xiB;
  _Float16 *xu16 = xu16A, *xi16 = xi16A, *xu16n = xu16B, *xi16n = xi16B;
  for (int l = 0; l < LAYERS; ++l) {
    ConvP pui = getConv(0, l);
    ConvP piu = getConv(1, l);
    // new_i = conv_ui(src=xu, dst=xi); new_u = conv_iu(src=xi, dst=xu)
    runConv(pui, 0 * LAYERS + l, xu16, xi16, xi, eamui16, ei_ui, xin, xi16n, isum);
    runConv(piu, 1 * LAYERS + l, xi16, xu16, xu, eamiu16, ei_iu, xun, xu16n, usum);
    { float* t = xu; xu = xun; xun = t; }  { float* t = xi; xi = xin; xin = t; }
    { _Float16* t = xu16; xu16 = xu16n; xu16n = t; }
    { _Float16* t = xi16; xi16 = xi16n; xi16n = t; }
  }

  // ---- head: final_cat -> fc1(+relu, WMMA) -> fc2 dot ----
  finalcat_kernel<<<gElem, blk, 0, stream>>>(usum, isum, easum, fin16, NB * 64);
  gemm64_kernel<0,1><<<gGemm, blk, 0, stream>>>(fin16, nullptr, packs + 20 * 4096, fc1_b, Hbuf, NB);
  fc2_kernel<<<gNode, blk, 0, stream>>>(Hbuf, fc2_w, fc2_b, (float*)d_out, NB);
}
